// FP8Linear_43980465111540
// MI455X (gfx1250) — compile-verified, matrix operand order checked by reading, not run on hardware
//
#include <hip/hip_runtime.h>
#include <stdint.h>

typedef __attribute__((ext_vector_type(16))) int   v16i;
typedef __attribute__((ext_vector_type(8)))  float v8f;
typedef int v4i_gcc __attribute__((vector_size(16)));   // matches async builtin param type

#define M_DIM 8192          // B*S = 2*4096
#define K_DIM 2048
#define N_DIM 8192
#define KT    (K_DIM / 128) // 16 k-tiles of depth 128
#define TILE_B 2048         // bytes of one packed 16x128 fp8 fragment tile
#define STAGE_A 32768       // 16 A tiles per stage (256 rows)
#define STAGE_BN 16384      // 8 B tiles per stage (128 cols)
#define Q_BYTES ((size_t)M_DIM * (size_t)K_DIM)   // 16 MiB packed A
#define FP8_MAXF 448.0f

// Byte offset of dword j (0..15), wave-half h (0..1) inside a 16x128 fp8
// fragment's K dimension (CDNA5 ISA §7.12.2).
__device__ __forceinline__ int kofs(int j, int h) {
    const int jj = j & 7;
    return ((j >> 3) << 6) + ((jj >> 1) << 4) + ((jj & 1) << 2) + (h << 3);
}

// ---------------- fp8 e4m3 conversion ----------------
__device__ __forceinline__ uint8_t sw_f32_to_e4m3(float f) {
    const uint8_t sign = (__float_as_uint(f) >> 24) & 0x80;
    float a = fminf(fabsf(f), FP8_MAXF);
    if (!(a >= 9.765625e-4f)) {
        int mant = (int)(a * 512.0f + 0.5f);
        if (mant <= 7) return sign | (uint8_t)mant;
        a = 1.953125e-3f * (float)mant;
    }
    int e;
    float mfrac = frexpf(a, &e);
    int exp = e - 1 + 7;
    int mant = (int)(mfrac * 16.0f + 0.5f);
    if (mant == 16) { mant = 8; exp++; }
    if (exp > 15 || (exp == 15 && (mant & 7) > 6)) return sign | 0x7E;
    if (exp < 1) return sign;
    return sign | (uint8_t)((exp << 3) | (mant & 7));
}

__device__ __forceinline__ float clamp8(float v, float inv) {
    return fminf(fmaxf(v * inv, -FP8_MAXF), FP8_MAXF);
}

__device__ __forceinline__ uint32_t pack4_e4m3(float a, float b, float c, float d) {
#if __has_builtin(__builtin_amdgcn_cvt_pk_fp8_f32)
    int p = __builtin_amdgcn_cvt_pk_fp8_f32(a, b, 0, false);
    p     = __builtin_amdgcn_cvt_pk_fp8_f32(c, d, p, true);
    return (uint32_t)p;
#else
    return (uint32_t)sw_f32_to_e4m3(a)
         | ((uint32_t)sw_f32_to_e4m3(b) << 8)
         | ((uint32_t)sw_f32_to_e4m3(c) << 16)
         | ((uint32_t)sw_f32_to_e4m3(d) << 24);
#endif
}

// ---------------- async global->LDS copy helpers ----------------
#if __has_builtin(__builtin_amdgcn_global_load_async_to_lds_b128)
#define USE_ASYNC 1
typedef __attribute__((address_space(1))) v4i_gcc* gbl_v4i_ptr;
typedef __attribute__((address_space(3))) v4i_gcc* lds_v4i_ptr;
#endif

__device__ __forceinline__ void cp16(const uint8_t* g, uint8_t* l) {
#ifdef USE_ASYNC
    __builtin_amdgcn_global_load_async_to_lds_b128(
        (gbl_v4i_ptr)g, (lds_v4i_ptr)l, 0, 0);
#else
    *(uint4*)l = *(const uint4*)g;
#endif
}

#ifdef USE_ASYNC
#define WAIT_ASYNC_12() asm volatile("s_wait_asynccnt 0xc" ::: "memory")
#define WAIT_ASYNC_0()  asm volatile("s_wait_asynccnt 0x0" ::: "memory")
#else
#define WAIT_ASYNC_12()
#define WAIT_ASYNC_0()
#endif

// ---------------- kernel 1: quantize x and pack into A-fragment layout ----------------
// One thread per PAIR of packed dwords: kofs(2t,h), kofs(2t+1,h) are contiguous,
// so each thread reads 8 consecutive floats (32B, full cacheline use) and stores
// a coalesced uint2.
__global__ void __launch_bounds__(256) quant_pack_a(const float* __restrict__ x,
                                                    const float* __restrict__ in_scale,
                                                    uint32_t* __restrict__ qpk) {
    const int tid  = blockIdx.x * blockDim.x + threadIdx.x;   // M*K/8 threads
    const int jp   = tid & 7;             // dword pair -> j = jp*2
    const int l    = (tid >> 3) & 31;
    const int tile = tid >> 8;            // tm*KT + tk
    const int tm   = tile >> 4;
    const int tk   = tile & (KT - 1);
    const int m    = (tm << 4) + (l & 15);
    const int k    = (tk << 7) + kofs(jp << 1, l >> 4);

    const float inv = 1.0f / in_scale[0];
    const float4 x0 = *(const float4*)(x + (size_t)m * K_DIM + k);
    const float4 x1 = *(const float4*)(x + (size_t)m * K_DIM + k + 4);

    uint2 p;
    p.x = pack4_e4m3(clamp8(x0.x, inv), clamp8(x0.y, inv), clamp8(x0.z, inv), clamp8(x0.w, inv));
    p.y = pack4_e4m3(clamp8(x1.x, inv), clamp8(x1.y, inv), clamp8(x1.z, inv), clamp8(x1.w, inv));
    *(uint2*)(qpk + (size_t)tid * 2) = p;
}

// ---------------- kernel 2: transpose-pack weight (K,N) into B-fragment layout ----------------
// LDS-tiled transpose: block handles 128(k) x 64(n). Phase 1 reads coalesced
// dwords, transposes 4x4 byte squares in registers, stores dwords into LDS in
// fragment order. Phase 2 streams LDS -> coalesced b128 global stores.
__global__ void __launch_bounds__(256) pack_b(const uint8_t* __restrict__ w,
                                              uint32_t* __restrict__ bpk) {
    __shared__ uint32_t lt[2048];          // 8 KiB: 4 tiles x 512 dwords
    const int tk = blockIdx.x;             // 0..KT-1
    const int n0 = blockIdx.y * 64;

#pragma unroll
    for (int s = 0; s < 2; ++s) {
        const int sq = s * 256 + threadIdx.x;  // 4x4 square index, 0..511
        const int kg = sq >> 4;                // k group 0..31 (k_local = kg*4)
        const int c  = sq & 15;                // n dword col 0..15
        const int kk = kg << 2;
        uint32_t d[4];
#pragma unroll
        for (int r = 0; r < 4; ++r)            // 4 rows, coalesced across lanes
            d[r] = *(const uint32_t*)(w + (size_t)(tk * 128 + kk + r) * N_DIM + n0 + c * 4);

        // fragment slot for k_local = kk (b spans 0..3 inside the dword)
        const int h     = (kk >> 3) & 1;
        const int jj    = (((kk >> 4) & 3) << 1) | ((kk >> 2) & 1);
        const int j     = (((kk >> 6) & 1) << 3) | jj;
        const int tile  = c >> 2;              // which of 4 B tiles
        const int lbase = ((c & 3) << 2) + (h << 4);
#pragma unroll
        for (int q = 0; q < 4; ++q) {          // byte-transpose 4x4 -> lane dwords
            const uint32_t o = ((d[0] >> (8 * q)) & 0xffu)
                             | (((d[1] >> (8 * q)) & 0xffu) << 8)
                             | (((d[2] >> (8 * q)) & 0xffu) << 16)
                             | (((d[3] >> (8 * q)) & 0xffu) << 24);
            lt[tile * 512 + (lbase + q) * 16 + j] = o;
        }
    }
    __syncthreads();

#pragma unroll
    for (int i = 0; i < 2; ++i) {              // 512 uint4 total, 2 per thread
        const int idx  = i * 256 + threadIdx.x;
        const int tile = idx >> 7;             // 128 uint4 per tile
        const int dd   = idx & 127;
        const uint4 v  = *(const uint4*)&lt[idx * 4];
        *(uint4*)((uint8_t*)bpk +
                  (size_t)((blockIdx.y * 4 + tile) * KT + tk) * TILE_B + dd * 16) = v;
    }
}

// ---------------- kernel 3: LDS-staged fp8 WMMA GEMM ----------------
// Block: 256 threads = 8 waves arranged 4(m) x 2(n); wave tile 64x64;
// block tile 256x128. Double-buffered (32K A + 16K B) stages, async filled.
// Fragment LDS traffic: 8 frags per 16 WMMAs = 1KB/WMMA per wave.
__global__ void __launch_bounds__(256) fp8_gemm(const uint8_t* __restrict__ Ag,
                                                const uint8_t* __restrict__ Bg,
                                                const float* __restrict__ bias,
                                                const float* __restrict__ s_in,
                                                const float* __restrict__ s_w,
                                                float* __restrict__ out) {
    extern __shared__ __align__(128) uint8_t smem[];   // 96 KiB: A[2][32K] B[2][16K]
    uint8_t* sA = smem;
    uint8_t* sB = smem + 2 * STAGE_A;

    const int lane  = threadIdx.x & 31;
    const int wave  = threadIdx.x >> 5;
    const int waveM = wave & 3;           // 0..3 -> 64 rows each
    const int waveN = wave >> 2;          // 0..1 -> 64 cols each
    const int tmb0  = blockIdx.x * 16;    // first 16-row subtile of block
    const int tnb0  = blockIdx.y * 8;     // first 16-col subtile of block

    // Each wave copies A tiles {wave, wave+8} and B tile {wave}: 12 async b128 ops.
    auto copy_stage = [&](int buf, int tk) {
        const uint8_t* ga0 = Ag + ((size_t)(tmb0 + wave)     * KT + tk) * TILE_B + lane * 64;
        const uint8_t* ga1 = Ag + ((size_t)(tmb0 + wave + 8) * KT + tk) * TILE_B + lane * 64;
        const uint8_t* gb  = Bg + ((size_t)(tnb0 + wave)     * KT + tk) * TILE_B + lane * 64;
        uint8_t* la0 = sA + buf * STAGE_A  + wave       * TILE_B + lane * 64;
        uint8_t* la1 = sA + buf * STAGE_A  + (wave + 8) * TILE_B + lane * 64;
        uint8_t* lb  = sB + buf * STAGE_BN + wave       * TILE_B + lane * 64;
#pragma unroll
        for (int i = 0; i < 4; ++i) {
            cp16(ga0 + 16 * i, la0 + 16 * i);
            cp16(ga1 + 16 * i, la1 + 16 * i);
            cp16(gb  + 16 * i, lb  + 16 * i);
        }
    };

    v8f acc[4][4] = {};
    copy_stage(0, 0);

    for (int tk = 0; tk < KT; ++tk) {
        const int cur = tk & 1;
        if (tk + 1 < KT) { copy_stage(cur ^ 1, tk + 1); WAIT_ASYNC_12(); }
        else             { WAIT_ASYNC_0(); }
        __syncthreads();                       // stage `cur` visible to all waves

        v16i b[4];
#pragma unroll
        for (int ni = 0; ni < 4; ++ni)
            b[ni] = *(const v16i*)(sB + cur * STAGE_BN + (waveN * 4 + ni) * TILE_B + lane * 64);

#pragma unroll
        for (int mi = 0; mi < 4; ++mi) {
            const v16i a = *(const v16i*)(sA + cur * STAGE_A + (waveM * 4 + mi) * TILE_B + lane * 64);
#pragma unroll
            for (int ni = 0; ni < 4; ++ni)
                acc[mi][ni] = __builtin_amdgcn_wmma_f32_16x16x128_fp8_fp8(
                    a, b[ni], (short)0, acc[mi][ni], false, false);
        }

        __syncthreads();                       // reads done before buffer reuse
    }

    const float s  = s_in[0] * s_w[0];
    const int   h  = lane >> 4;
    const int   nl = lane & 15;
#pragma unroll
    for (int mi = 0; mi < 4; ++mi)
#pragma unroll
        for (int ni = 0; ni < 4; ++ni) {
            const int   ng = (tnb0 + waveN * 4 + ni) * 16 + nl;
            const float bv = bias[ng];
#pragma unroll
            for (int r = 0; r < 8; ++r) {
                const int mg = (tmb0 + waveM * 4 + mi) * 16 + h * 8 + r;
                out[(size_t)mg * N_DIM + ng] = acc[mi][ni][r] * s + bv;
            }
        }
}

extern "C" void kernel_launch(void* const* d_in, const int* in_sizes, int n_in,
                              void* d_out, int out_size, void* d_ws, size_t ws_size,
                              hipStream_t stream) {
    const float*   x    = (const float*)d_in[0];
    const uint8_t* w    = (const uint8_t*)d_in[1];   // fp8 e4m3 bytes
    const float*   w_s  = (const float*)d_in[2];
    const float*   in_s = (const float*)d_in[3];
    const float*   bias = (const float*)d_in[4];
    float*         out  = (float*)d_out;

    uint32_t* qpk = (uint32_t*)d_ws;                           // 16 MiB packed A
    uint32_t* bpk = (uint32_t*)((char*)d_ws + Q_BYTES);        // 16 MiB packed B

    const int nQ = (M_DIM * K_DIM / 8) / 256;   // 8192 blocks
    quant_pack_a<<<nQ, 256, 0, stream>>>(x, in_s, qpk);

    dim3 gridB(KT, N_DIM / 64);                 // 16 x 128 blocks
    pack_b<<<gridB, 256, 0, stream>>>(w, bpk);

    dim3 grid(M_DIM / 256, N_DIM / 128);        // 32 x 64 blocks
    fp8_gemm<<<grid, 256, 98304, stream>>>((const uint8_t*)qpk, (const uint8_t*)bpk,
                                           bias, in_s, w_s, out);
}
